// MsPAM_Module_38233798869171
// MI455X (gfx1250) — matrix operand order repaired and checked by reading, not compiled, not run on hardware
//
#include <hip/hip_runtime.h>

// ---------------------------------------------------------------------------
// MsPAM (multi-source position attention) for MI455X / gfx1250, wave32 WMMA.
// - All GEMMs: v_wmma_f32_16x16x32_f16 (f16 in, f32 accumulate).
// - Flash-style two-pass softmax (no 64MB score materialization).
// - Pass B stages shared V/Q slabs into LDS with the Tensor Data Mover
//   (tensor_load_to_lds), double-buffered, synced via s_wait_tensorcnt +
//   workgroup barriers.
// ---------------------------------------------------------------------------

typedef _Float16 h8   __attribute__((ext_vector_type(8)));
typedef _Float16 v16h __attribute__((ext_vector_type(16)));
typedef float    v8f  __attribute__((ext_vector_type(8)));
typedef unsigned int u32x4 __attribute__((ext_vector_type(4)));
typedef int      i32x4 __attribute__((ext_vector_type(4)));
typedef int      i32x8 __attribute__((ext_vector_type(8)));

#define CDIM 512
#define CQDIM 64
#define NPIX 4096
#define BATCH 2

static __device__ __forceinline__ v8f wmma16(v16h a, v16h b, v8f c) {
  // D = A(16x32 f16) * B(32x16 f16) + C(16x16 f32)
  return __builtin_amdgcn_wmma_f32_16x16x32_f16(false, a, false, b, (short)0, c,
                                                false, false);
}

// A-operand tile (16 rows x 32 K, f16) from a row-major [rows x stride] half
// matrix (global or LDS; addrspace inferred after inlining). Contiguous K order (as in the B loaders):
// lanes 0-15 hold row M=l, K in {0..15}; lanes 16-31 hold row M=l-16,
// K in {16..31}.
static __device__ __forceinline__ v16h load_A16(const _Float16* base, int stride,
                                                int row0, int k0, int lane) {
  const int lq = lane & 15, hf = lane >> 4;
  const _Float16* p = base + (size_t)(row0 + lq) * stride + (k0 + hf * 16);
  h8 lo = *(const h8*)p;
  h8 hi = *(const h8*)(p + 8);
  v16h r;
#pragma unroll
  for (int i = 0; i < 8; ++i) { r[i] = lo[i]; r[i + 8] = hi[i]; }
  return r;
}

// B-operand tile (32 K x 16 cols, f16) from a TRANSPOSED half matrix
// stored [cols x stride] so each lane reads 16 contiguous halves.
static __device__ __forceinline__ v16h load_B16T(const _Float16* baseT, int stride,
                                                 int col0, int k0, int lane) {
  const int lq = lane & 15, hf = lane >> 4;
  const _Float16* p = baseT + (size_t)(col0 + lq) * stride + (k0 + hf * 16);
  h8 lo = *(const h8*)p;
  h8 hi = *(const h8*)(p + 8);
  v16h r;
#pragma unroll
  for (int i = 0; i < 8; ++i) { r[i] = lo[i]; r[i + 8] = hi[i]; }
  return r;
}

// B-operand tile gathered from an fp32 row-major [K x ldn] activation matrix
// (strided column gather + cvt to f16). Used only in the projection GEMMs.
static __device__ __forceinline__ v16h load_B16g(const float* X, int ldn,
                                                 int k0, int col0, int lane) {
  const int lq = lane & 15, hf = lane >> 4;
  const float* p = X + (size_t)(k0 + hf * 16) * ldn + (col0 + lq);
  v16h r;
#pragma unroll
  for (int i = 0; i < 16; ++i) r[i] = (_Float16)p[(size_t)i * ldn];
  return r;
}

// ---------------------------------------------------------------------------
// Tensor Data Mover: 2D f16 tile (tile_d1 rows x tile_d0 elems) from global
// (row stride = stride_d0 elems) into contiguous LDS at lds_off.
// D# packing per CDNA5 ISA.
static __device__ __forceinline__ void tdm_load_2d_f16(
    unsigned lds_off, const void* gaddr, unsigned tensor_d0, unsigned tensor_d1,
    unsigned tile_d0, unsigned tile_d1, unsigned stride_d0) {
  const size_t ga = (size_t)gaddr;
  u32x4 g0;
  g0[0] = 1u;                                              // count=1 (user D#)
  g0[1] = lds_off;                                         // lds_addr
  g0[2] = (unsigned)(ga & 0xffffffffu);                    // global_addr lo
  g0[3] = (unsigned)((ga >> 32) & 0x01ffffffu) | (2u << 30);  // hi | type=2
  i32x8 g1;
  g1[0] = (int)(1u << 16);                                 // data_size=2B
  g1[1] = (int)((tensor_d0 & 0xffffu) << 16);              // tensor_dim0 lo16
  g1[2] = (int)(((tensor_d0 >> 16) & 0xffffu) | ((tensor_d1 & 0xffffu) << 16));
  g1[3] = (int)(((tensor_d1 >> 16) & 0xffffu) | ((tile_d0 & 0xffffu) << 16));
  g1[4] = (int)(tile_d1 & 0xffffu);                        // tile_dim1; dim2=0
  g1[5] = (int)stride_d0;                                  // dim0_stride lo32
  g1[6] = 0;                                               // stride hi; dim1_stride
  g1[7] = 0;
  i32x4 g2 = {};
  i32x4 g3 = {};
#if defined(__clang_major__) && __clang_major__ >= 23
  i32x8 g1b = {};
  __builtin_amdgcn_tensor_load_to_lds(g0, g1, g2, g3, g1b, 0);
#else
  __builtin_amdgcn_tensor_load_to_lds(g0, g1, g2, g3, 0);
#endif
}

static __device__ __forceinline__ unsigned lds_off_u32(const void* p) {
  return (unsigned)(size_t)p;  // LDS address = low 32 bits of the generic pointer
}

// ---------------------------------------------------------------------------
__global__ void cvt_f32_to_f16(const float* __restrict__ src,
                               _Float16* __restrict__ dst, int n) {
  int i = blockIdx.x * blockDim.x + threadIdx.x;
  if (i < n) dst[i] = (_Float16)src[i];
}

// ---------------------------------------------------------------------------
// Projection: out = W[M x 512] * X[512 x 4096] + bias, per batch (blockIdx.z).
template <int MT>
__global__ void __launch_bounds__(256) proj_kernel(
    const _Float16* __restrict__ W, const float* __restrict__ bias,
    const float* __restrict__ X, _Float16* __restrict__ outT,
    _Float16* __restrict__ outN, int M) {
  const int lane = threadIdx.x & 31;
  const int wave = threadIdx.x >> 5;
  const int lq = lane & 15, hf = lane >> 4;
  const int n0 = (blockIdx.x * 8 + wave) * 16;
  const int m0 = blockIdx.y * (MT * 16);
  const int b = blockIdx.z;
  const float* Xb = X + (size_t)b * CDIM * NPIX;

  v8f acc[MT] = {};
  for (int k0 = 0; k0 < CDIM; k0 += 32) {
    v16h bmat = load_B16g(Xb, NPIX, k0, n0, lane);
#pragma unroll
    for (int t = 0; t < MT; ++t) {
      v16h amat = load_A16(W, CDIM, m0 + t * 16, k0, lane);
      acc[t] = wmma16(amat, bmat, acc[t]);
    }
  }
#pragma unroll
  for (int t = 0; t < MT; ++t) {
    const int mrow = m0 + t * 16 + hf * 8;
    const float* bp = bias + mrow;
    h8 hv;
#pragma unroll
    for (int r = 0; r < 8; ++r) hv[r] = (_Float16)(acc[t][r] + bp[r]);
    if (outT) *(h8*)(outT + ((size_t)b * NPIX + n0 + lq) * M + mrow) = hv;
    if (outN) {
      _Float16* q = outN + ((size_t)b * M + mrow) * NPIX + (n0 + lq);
#pragma unroll
      for (int r = 0; r < 8; ++r) q[(size_t)r * NPIX] = hv[r];
    }
  }
}

// ---------------------------------------------------------------------------
// Pass A: per (branch, batch, 16-row i-tile), online softmax row stats of
// S[i,:] = Q(:,i).K(:,j) via WMMA score tiles.
__global__ void __launch_bounds__(256) passA_kernel(
    const _Float16* __restrict__ Qt, const _Float16* __restrict__ K1t,
    const _Float16* __restrict__ K2t, const _Float16* __restrict__ K3t,
    float* __restrict__ rowmax, float* __restrict__ rowsum) {
  const int lane = threadIdx.x & 31;
  const int wave = threadIdx.x >> 5;
  const int lq = lane & 15, hf = lane >> 4;
  const int i0 = (blockIdx.x * 8 + wave) * 16;
  const int br = blockIdx.y;
  const int b = blockIdx.z;
  const _Float16* Qb = Qt + (size_t)b * NPIX * CQDIM;
  const _Float16* Kt =
      (br == 0 ? K1t : (br == 1 ? K2t : K3t)) + (size_t)b * NPIX * CQDIM;

  v16h a0 = load_A16(Qb, CQDIM, i0, 0, lane);
  v16h a1 = load_A16(Qb, CQDIM, i0, 32, lane);

  float m[8], l[8];
#pragma unroll
  for (int r = 0; r < 8; ++r) { m[r] = -3.0e38f; l[r] = 0.f; }

  for (int j0 = 0; j0 < NPIX; j0 += 16) {
    if (j0 + 64 < NPIX)
      __builtin_prefetch(Kt + (size_t)(j0 + 64) * CQDIM, 0, 1);
    v16h b0 = load_B16T(Kt, CQDIM, j0, 0, lane);
    v16h b1 = load_B16T(Kt, CQDIM, j0, 32, lane);
    v8f s = {};
    s = wmma16(a0, b0, s);
    s = wmma16(a1, b1, s);
#pragma unroll
    for (int r = 0; r < 8; ++r) {
      float x = s[r];
      float mn = fmaxf(m[r], x);
      l[r] = l[r] * __expf(m[r] - mn) + __expf(x - mn);
      m[r] = mn;
    }
  }
#pragma unroll
  for (int off = 1; off < 16; off <<= 1) {
#pragma unroll
    for (int r = 0; r < 8; ++r) {
      float mo = __shfl_xor(m[r], off);
      float lo = __shfl_xor(l[r], off);
      float mn = fmaxf(m[r], mo);
      l[r] = l[r] * __expf(m[r] - mn) + lo * __expf(mo - mn);
      m[r] = mn;
    }
  }
  if (lq == 0) {
    float* mp = rowmax + ((size_t)br * BATCH + b) * NPIX + i0 + hf * 8;
    float* lp = rowsum + ((size_t)br * BATCH + b) * NPIX + i0 + hf * 8;
#pragma unroll
    for (int r = 0; r < 8; ++r) { mp[r] = m[r]; lp[r] = l[r]; }
  }
}

// ---------------------------------------------------------------------------
// Pass B: att(c,n) = sum_br sum_i P_br[i,n]*V[c,i]; P = exp(S-m_i)/l_i with S
// recomputed tile-wise. Block: 8 waves sharing c-chunk (128 rows of V) and
// the i loop; V and Q slabs are TDM-staged into LDS once per block, double
// buffered two chunks deep. Per-wave P tiles route via LDS transpose.
__global__ void __launch_bounds__(256) passB_kernel(
    const _Float16* __restrict__ Qt, const _Float16* __restrict__ K1t,
    const _Float16* __restrict__ K2t, const _Float16* __restrict__ K3t,
    const _Float16* __restrict__ V, const float* __restrict__ rowmax,
    const float* __restrict__ rowsum, _Float16* __restrict__ attT) {
  __shared__ __align__(16) _Float16 vbuf[2][128][32];  // V slab  [c_loc][i_loc]
  __shared__ __align__(16) _Float16 qbuf[2][32][64];   // Q slab  [i_loc][c_q]
  __shared__ __align__(16) _Float16 pbuf[8][16][32];   // P tiles [wave][n][i]
  const int lane = threadIdx.x & 31;
  const int wave = threadIdx.x >> 5;
  const int lq = lane & 15, hf = lane >> 4;
  const int n0 = (blockIdx.x * 8 + wave) * 16;
  const int c0 = blockIdx.y * 128;
  const int b = blockIdx.z;
  const _Float16* Qb = Qt + (size_t)b * NPIX * CQDIM;
  const _Float16* Vb = V + (size_t)b * CDIM * NPIX;
  const int NCH = NPIX / 32;  // 128 i-chunks

  // Hoisted per-branch score B-operands (depend only on the n-tile).
  v16h bk0[3], bk1[3];
  const float *mrowb[3], *lrowb[3];
#pragma unroll
  for (int br = 0; br < 3; ++br) {
    const _Float16* Kt =
        (br == 0 ? K1t : (br == 1 ? K2t : K3t)) + (size_t)b * NPIX * CQDIM;
    bk0[br] = load_B16T(Kt, CQDIM, n0, 0, lane);
    bk1[br] = load_B16T(Kt, CQDIM, n0, 32, lane);
    mrowb[br] = rowmax + ((size_t)br * BATCH + b) * NPIX;
    lrowb[br] = rowsum + ((size_t)br * BATCH + b) * NPIX;
  }

  // TDM pipeline prologue: chunks 0 and 1 (2 ops each: V slab + Q slab).
  if (wave == 0) {
#pragma unroll
    for (int pc = 0; pc < 2; ++pc) {
      tdm_load_2d_f16(lds_off_u32(&vbuf[pc][0][0]),
                      Vb + (size_t)c0 * NPIX + pc * 32,
                      NPIX, CDIM, 32, 128, NPIX);
      tdm_load_2d_f16(lds_off_u32(&qbuf[pc][0][0]),
                      Qb + (size_t)(pc * 32) * CQDIM,
                      CQDIM, NPIX, 64, 32, CQDIM);
    }
  }

  v8f acc[8] = {};

  for (int ch = 0; ch < NCH; ++ch) {
    const int i0 = ch * 32;
    const int cur = ch & 1;
    // Oldest in-flight pair (this chunk) must be complete. Waiting with
    // TENSORcnt==0 on non-issuing waves is a no-op.
    if (ch + 1 < NCH) __builtin_amdgcn_s_wait_tensorcnt(2);
    else              __builtin_amdgcn_s_wait_tensorcnt(0);
    __syncthreads();  // publish TDM-written LDS to all waves

    // Q score A-operands from LDS: shared by all 3 branches.
    v16h qa00 = load_A16(&qbuf[cur][0][0], CQDIM, 0, 0, lane);
    v16h qa01 = load_A16(&qbuf[cur][0][0], CQDIM, 0, 32, lane);
    v16h qa10 = load_A16(&qbuf[cur][0][0], CQDIM, 16, 0, lane);
    v16h qa11 = load_A16(&qbuf[cur][0][0], CQDIM, 16, 32, lane);

#pragma unroll
    for (int br = 0; br < 3; ++br) {
#pragma unroll
      for (int it = 0; it < 2; ++it) {
        v8f s = {};
        s = wmma16(it ? qa10 : qa00, bk0[br], s);
        s = wmma16(it ? qa11 : qa01, bk1[br], s);
        // D layout: lane holds rows (i0+it*16+hf*8+r), column n0+lq.
        const float* mp = mrowb[br] + i0 + it * 16 + hf * 8;
        const float* lp = lrowb[br] + i0 + it * 16 + hf * 8;
        h8 pv;
#pragma unroll
        for (int r = 0; r < 8; ++r)
          pv[r] = (_Float16)(__expf(s[r] - mp[r]) / lp[r]);
        *(h8*)&pbuf[wave][lq][it * 16 + hf * 8] = pv;  // LDS transpose
      }
      asm volatile("s_wait_dscnt 0" ::: "memory");
      h8 plo = *(const h8*)&pbuf[wave][lq][hf * 16];
      h8 phi = *(const h8*)&pbuf[wave][lq][hf * 16 + 8];
      v16h bp;
#pragma unroll
      for (int i = 0; i < 8; ++i) { bp[i] = plo[i]; bp[i + 8] = phi[i]; }
#pragma unroll
      for (int t = 0; t < 8; ++t) {
        v16h av = load_A16(&vbuf[cur][0][0], 32, t * 16, 0, lane);
        acc[t] = wmma16(av, bp, acc[t]);
      }
    }
    __syncthreads();  // all waves done with buf[cur] before refill
    if (wave == 0 && ch + 2 < NCH) {
      tdm_load_2d_f16(lds_off_u32(&vbuf[cur][0][0]),
                      Vb + (size_t)c0 * NPIX + (size_t)(ch + 2) * 32,
                      NPIX, CDIM, 32, 128, NPIX);
      tdm_load_2d_f16(lds_off_u32(&qbuf[cur][0][0]),
                      Qb + (size_t)((ch + 2) * 32) * CQDIM,
                      CQDIM, NPIX, 64, 32, CQDIM);
    }
  }
  __builtin_amdgcn_s_wait_tensorcnt(0);  // drain before exit

#pragma unroll
  for (int t = 0; t < 8; ++t) {
    h8 hv;
#pragma unroll
    for (int r = 0; r < 8; ++r) hv[r] = (_Float16)acc[t][r];
    *(h8*)(attT + ((size_t)b * NPIX + n0 + lq) * CDIM + c0 + t * 16 + hf * 8) =
        hv;
  }
}

// ---------------------------------------------------------------------------
// Final: out = x + gamma * (Wl @ att + bl), fp32 output.
__global__ void __launch_bounds__(256) final_kernel(
    const _Float16* __restrict__ Wl, const float* __restrict__ bl,
    const _Float16* __restrict__ attT, const float* __restrict__ x,
    const float* __restrict__ gamma, float* __restrict__ out) {
  const int lane = threadIdx.x & 31;
  const int wave = threadIdx.x >> 5;
  const int lq = lane & 15, hf = lane >> 4;
  const int n0 = (blockIdx.x * 8 + wave) * 16;
  const int m0 = blockIdx.y * 128;
  const int b = blockIdx.z;
  const _Float16* Ab = attT + (size_t)b * NPIX * CDIM;
  const float g = gamma[0];

  v8f acc[8] = {};
  for (int k0 = 0; k0 < CDIM; k0 += 32) {
    v16h bmat = load_B16T(Ab, CDIM, n0, k0, lane);
#pragma unroll
    for (int t = 0; t < 8; ++t) {
      v16h amat = load_A16(Wl, CDIM, m0 + t * 16, k0, lane);
      acc[t] = wmma16(amat, bmat, acc[t]);
    }
  }
#pragma unroll
  for (int t = 0; t < 8; ++t) {
    const int mrow = m0 + t * 16 + hf * 8;
    const float* bp = bl + mrow;
    const float* xp = x + ((size_t)b * CDIM + mrow) * NPIX + n0 + lq;
    float* op = out + ((size_t)b * CDIM + mrow) * NPIX + n0 + lq;
#pragma unroll
    for (int r = 0; r < 8; ++r)
      op[(size_t)r * NPIX] = xp[(size_t)r * NPIX] + g * (acc[t][r] + bp[r]);
  }
}

// ---------------------------------------------------------------------------
extern "C" void kernel_launch(void* const* d_in, const int* in_sizes, int n_in,
                              void* d_out, int out_size, void* d_ws,
                              size_t ws_size, hipStream_t stream) {
  const float* x = (const float*)d_in[0];
  const float* y = (const float*)d_in[1];
  const float* z = (const float*)d_in[2];
  const float* Wq = (const float*)d_in[3];
  const float* bq = (const float*)d_in[4];
  const float* Wk1 = (const float*)d_in[5];
  const float* bk1 = (const float*)d_in[6];
  const float* Wk2 = (const float*)d_in[7];
  const float* bk2 = (const float*)d_in[8];
  const float* Wk3 = (const float*)d_in[9];
  const float* bk3 = (const float*)d_in[10];
  const float* Wv = (const float*)d_in[11];
  const float* bv = (const float*)d_in[12];
  const float* Wl = (const float*)d_in[13];
  const float* bl = (const float*)d_in[14];
  const float* gamma = (const float*)d_in[15];
  float* out = (float*)d_out;

  char* ws = (char*)d_ws;
  size_t off = 0;
  auto alloc = [&](size_t bytes) -> char* {
    char* p = ws + off;
    off = (off + bytes + 255) & ~(size_t)255;
    return p;
  };
  _Float16* Wqh = (_Float16*)alloc((size_t)CQDIM * CDIM * 2);
  _Float16* Wk1h = (_Float16*)alloc((size_t)CQDIM * CDIM * 2);
  _Float16* Wk2h = (_Float16*)alloc((size_t)CQDIM * CDIM * 2);
  _Float16* Wk3h = (_Float16*)alloc((size_t)CQDIM * CDIM * 2);
  _Float16* Wvh = (_Float16*)alloc((size_t)CDIM * CDIM * 2);
  _Float16* Wlh = (_Float16*)alloc((size_t)CDIM * CDIM * 2);
  _Float16* QtW = (_Float16*)alloc((size_t)BATCH * NPIX * CQDIM * 2);
  _Float16* K1t = (_Float16*)alloc((size_t)BATCH * NPIX * CQDIM * 2);
  _Float16* K2t = (_Float16*)alloc((size_t)BATCH * NPIX * CQDIM * 2);
  _Float16* K3t = (_Float16*)alloc((size_t)BATCH * NPIX * CQDIM * 2);
  _Float16* Vh = (_Float16*)alloc((size_t)BATCH * CDIM * NPIX * 2);
  _Float16* attT = (_Float16*)alloc((size_t)BATCH * NPIX * CDIM * 2);
  float* rowmax = (float*)alloc((size_t)3 * BATCH * NPIX * 4);
  float* rowsum = (float*)alloc((size_t)3 * BATCH * NPIX * 4);
  if (off > ws_size) return;  // scratch too small: do nothing (deterministic)

  // 1) weight conversion f32 -> f16
  const int nq = CQDIM * CDIM, nc = CDIM * CDIM;
  cvt_f32_to_f16<<<(nq + 255) / 256, 256, 0, stream>>>(Wq, Wqh, nq);
  cvt_f32_to_f16<<<(nq + 255) / 256, 256, 0, stream>>>(Wk1, Wk1h, nq);
  cvt_f32_to_f16<<<(nq + 255) / 256, 256, 0, stream>>>(Wk2, Wk2h, nq);
  cvt_f32_to_f16<<<(nq + 255) / 256, 256, 0, stream>>>(Wk3, Wk3h, nq);
  cvt_f32_to_f16<<<(nc + 255) / 256, 256, 0, stream>>>(Wv, Wvh, nc);
  cvt_f32_to_f16<<<(nc + 255) / 256, 256, 0, stream>>>(Wl, Wlh, nc);

  // 2) projections (Q/K transposed for later B-operand reads, V natural)
  dim3 gq(32, 1, BATCH), gv(32, 4, BATCH);
  proj_kernel<4><<<gq, 256, 0, stream>>>(Wqh, bq, x, QtW, nullptr, CQDIM);
  proj_kernel<4><<<gq, 256, 0, stream>>>(Wk1h, bk1, x, K1t, nullptr, CQDIM);
  proj_kernel<4><<<gq, 256, 0, stream>>>(Wk2h, bk2, y, K2t, nullptr, CQDIM);
  proj_kernel<4><<<gq, 256, 0, stream>>>(Wk3h, bk3, z, K3t, nullptr, CQDIM);
  proj_kernel<8><<<gv, 256, 0, stream>>>(Wvh, bv, x, nullptr, Vh, CDIM);

  // 3) softmax row statistics (flash pass A), all 3 branches
  passA_kernel<<<dim3(32, 3, BATCH), 256, 0, stream>>>(QtW, K1t, K2t, K3t,
                                                       rowmax, rowsum);

  // 4) fused P*V accumulation over 3 branches (flash pass B, TDM-staged)
  passB_kernel<<<dim3(32, 4, BATCH), 256, 0, stream>>>(
      QtW, K1t, K2t, K3t, Vh, rowmax, rowsum, attT);

  // 5) final projection + residual + gamma
  final_kernel<<<dim3(32, 4, BATCH), 256, 0, stream>>>(Wlh, bl, attT, x, gamma,
                                                       out);
}
